// Block_ToMeATC_21689584844858
// MI455X (gfx1250) — matrix operand order repaired and checked by reading, not compiled, not run on hardware
//
#include <hip/hip_runtime.h>
#include <hip/hip_bf16.h>
#include <math.h>

// ---------------------------------------------------------------------------
// Problem constants (from reference): B=8, N=1024, C=768, H=12, HD=64
// TOKEN_RATIO=0.5 -> num_clusters=512, r = min(512,512) = 512, n_unm = 0,
// merged token count = N - r = 512.
// ---------------------------------------------------------------------------
constexpr int kB  = 8;
constexpr int kN  = 1024;
constexpr int kC  = 768;
constexpr int kH  = 12;
constexpr int kHD = 64;
constexpr int kR  = 512;           // merged src count
constexpr int kNU = (kN / 2) - kR; // unmerged count (= 0 here)
constexpr int kTD = kN - kR;       // output tokens (= 512)
constexpr float kEPS = 1e-5f;

typedef __attribute__((ext_vector_type(16))) __bf16 bf16x16;
typedef __attribute__((ext_vector_type(8)))  __bf16 bf16x8;
typedef __attribute__((ext_vector_type(8)))  float  floatx8;

// ---------------------------------------------------------------------------
// WMMA fragment helpers (wave32; A and B share a contiguous k order chosen here, not the ISA's table order):
// A 16x32 bf16: lanes 0-15 => rows M=0..15 with K=0..15,
//               lanes 16-31 => same rows with K=16..31
//   -> per lane: two contiguous 8-half (16B) chunks at +((hi)?16:0) and +8.
// B 32x16 bf16: lane n=lane&15 => output column; lanes 0-15 K=0..15,
//               lanes 16-31 K=16..31 -> one contiguous 16-half (32B) chunk.
// C/D 16x16 f32: lane n=lane&15 col, VGPR v => row v + (lane>=16 ? 8 : 0).
// ---------------------------------------------------------------------------
__device__ __forceinline__ bf16x16 load_a_frag(const __bf16* base, int ldk, int lane) {
  const int hi = (lane >= 16) ? 16 : 0;
  const int m  = lane & 15;
  const __bf16* p = base + (size_t)m * ldk + hi;
  bf16x8 c0 = *(const bf16x8*)(p);
  bf16x8 c1 = *(const bf16x8*)(p + 8);
  return __builtin_shufflevector(c0, c1, 0,1,2,3,4,5,6,7,8,9,10,11,12,13,14,15);
}

__device__ __forceinline__ bf16x16 load_b_frag(const __bf16* base, int ldk, int lane) {
  const int hi = (lane >= 16) ? 16 : 0;
  const int n  = lane & 15;
  return *(const bf16x16*)(base + (size_t)n * ldk + hi);
}

__device__ __forceinline__ floatx8 wmma_bf16(bf16x16 a, bf16x16 b, floatx8 c) {
  return __builtin_amdgcn_wmma_f32_16x16x32_bf16(false, a, false, b, (short)0, c, false, false);
}

// ---------------------------------------------------------------------------
// CDNA5 async global->LDS copy (ASYNCcnt-tracked, VGLOBAL encoding).
// LDS destination address = low 32 bits of the flat shared-aperture address
// (see ISA 10.2).
// ---------------------------------------------------------------------------
__device__ __forceinline__ void async_b128(unsigned int lds_off, const void* g) {
  unsigned long long ga = (unsigned long long)(uintptr_t)g;
  asm volatile("global_load_async_to_lds_b128 %0, %1, off"
               :: "v"(lds_off), "v"(ga) : "memory");
}
__device__ __forceinline__ void wait_async0() {
  asm volatile("s_wait_asynccnt 0x0" ::: "memory");
}
__device__ __forceinline__ unsigned int lds_off(const void* p) {
  return (unsigned int)(uintptr_t)p;
}

// ---------------------------------------------------------------------------
// fp32 -> bf16 conversion (weights)
// ---------------------------------------------------------------------------
__global__ void cvt_f32_to_bf16(const float* __restrict__ in, __bf16* __restrict__ out, int n) {
  int i = blockIdx.x * 256 + threadIdx.x;
  if (i < n) out[i] = (__bf16)in[i];
}

// ---------------------------------------------------------------------------
// LayerNorm (fp32 in, bf16 out). One block (256 threads) per token row.
// ---------------------------------------------------------------------------
__global__ void layernorm_to_bf16(const float* __restrict__ x,
                                  const float* __restrict__ w,
                                  const float* __restrict__ bvec,
                                  __bf16* __restrict__ out, int C) {
  const int row = blockIdx.x;
  const float* xr = x + (size_t)row * C;
  __shared__ float red[256];
  float s = 0.f;
  for (int c = threadIdx.x; c < C; c += 256) s += xr[c];
  red[threadIdx.x] = s; __syncthreads();
  for (int off = 128; off > 0; off >>= 1) {
    if (threadIdx.x < off) red[threadIdx.x] += red[threadIdx.x + off];
    __syncthreads();
  }
  const float mu = red[0] / (float)C;
  __syncthreads();
  float v = 0.f;
  for (int c = threadIdx.x; c < C; c += 256) { float d = xr[c] - mu; v += d * d; }
  red[threadIdx.x] = v; __syncthreads();
  for (int off = 128; off > 0; off >>= 1) {
    if (threadIdx.x < off) red[threadIdx.x] += red[threadIdx.x + off];
    __syncthreads();
  }
  const float inv = rsqrtf(red[0] / (float)C + kEPS);
  for (int c = threadIdx.x; c < C; c += 256)
    out[(size_t)row * C + c] = (__bf16)((xr[c] - mu) * inv * w[c] + bvec[c]);
}

// ---------------------------------------------------------------------------
// Generic bf16 WMMA GEMM: C[M,N] = A[M,K] @ W[N,K]^T  (+ epilogue)
//   mode 0: store bf16 (no bias)          -> outH
//   mode 1: + bias[n] + resid[m,n], fp32  -> outF
//   mode 2: gelu(acc + bias[n]), bf16     -> outH
// Block = 128 threads = 4 waves; block tile 64x128 (M x N); each wave owns a
// 64x32 strip (4x2 WMMA tiles). K staged in 32-wide steps through
// double-buffered LDS filled with global_load_async_to_lds_b128 so the loads
// for step t+1 overlap the WMMAs of step t (one s_wait_asynccnt + barrier per
// step). K % 32 == 0, M % 64 == 0, N % 128 == 0.
// ---------------------------------------------------------------------------
__global__ void gemm_bf16_nt(const __bf16* __restrict__ A, const __bf16* __restrict__ W,
                             const float* __restrict__ bias, const float* __restrict__ resid,
                             float* __restrict__ outF, __bf16* __restrict__ outH,
                             int M, int N, int K, int mode) {
  __shared__ __align__(64) __bf16 As[2][64 * 32];
  __shared__ __align__(64) __bf16 Bs[2][128 * 32];
  const int n0 = blockIdx.x * 128, m0 = blockIdx.y * 64;
  const int tid = threadIdx.x, lane = tid & 31, wv = tid >> 5;
  const int nq = wv * 32;   // wave's 32 output columns within the 128-wide tile

  // staging coordinates: A -> thread covers row tid>>1, 16 elements; B -> row tid.
  const int ar = tid >> 1;
  const int ac = (tid & 1) * 16;
  const __bf16* gA = A + (size_t)(m0 + ar) * K + ac;
  const __bf16* gB = W + (size_t)(n0 + tid) * K;
  unsigned int asoff[2], bsoff[2];
#pragma unroll
  for (int p = 0; p < 2; ++p) {
    asoff[p] = lds_off(&As[p][ar * 32 + ac]);
    bsoff[p] = lds_off(&Bs[p][tid * 32]);
  }
  auto stage = [&](int p, int kt) {
    async_b128(asoff[p],      gA + kt);
    async_b128(asoff[p] + 16, gA + kt + 8);
    async_b128(bsoff[p],      gB + kt);
    async_b128(bsoff[p] + 16, gB + kt + 8);
    async_b128(bsoff[p] + 32, gB + kt + 16);
    async_b128(bsoff[p] + 48, gB + kt + 24);
  };

  floatx8 acc[4][2] = {};
  const int nk = K / 32;
  stage(0, 0);
  for (int t = 0; t < nk; ++t) {
    const int p = t & 1;
    wait_async0();       // my async loads into buffer p are complete
    __syncthreads();     // everyone's loads done; everyone done reading buffer p^1
    if (t + 1 < nk) stage(p ^ 1, (t + 1) * 32);
    if (t + 2 < nk) {
      __builtin_prefetch(gA + (t + 2) * 32, 0, 1);
      __builtin_prefetch(gB + (t + 2) * 32, 0, 1);
    }
    bf16x16 bfr0 = load_b_frag(&Bs[p][(nq)      * 32], 32, lane);
    bf16x16 bfr1 = load_b_frag(&Bs[p][(nq + 16) * 32], 32, lane);
#pragma unroll
    for (int mi = 0; mi < 4; ++mi) {
      bf16x16 af = load_a_frag(&As[p][16 * mi * 32], 32, lane);
      acc[mi][0] = wmma_bf16(af, bfr0, acc[mi][0]);
      acc[mi][1] = wmma_bf16(af, bfr1, acc[mi][1]);
    }
  }

  const int hi = (lane >= 16) ? 8 : 0;
  const int nn = lane & 15;
#pragma unroll
  for (int mi = 0; mi < 4; ++mi)
#pragma unroll
    for (int tj = 0; tj < 2; ++tj) {
#pragma unroll
      for (int v = 0; v < 8; ++v) {
        const int gr = m0 + 16 * mi + v + hi;
        const int gc = n0 + nq + 16 * tj + nn;
        const float val = acc[mi][tj][v];
        if (mode == 0) {
          outH[(size_t)gr * N + gc] = (__bf16)val;
        } else if (mode == 1) {
          outF[(size_t)gr * N + gc] = val + bias[gc] + resid[(size_t)gr * N + gc];
        } else {
          float t2 = val + bias[gc];
          outH[(size_t)gr * N + gc] = (__bf16)(0.5f * t2 * (1.f + erff(t2 * 0.70710678118654752f)));
        }
      }
    }
}

// ---------------------------------------------------------------------------
// Fused attention for one (b, h, 16-query tile). Block = 128 threads (4 waves).
// qkv is bf16, layout [b, n, s*768 + h*64 + d] (s=0:q, 1:k, 2:v).
// Dynamic LDS: S[16][1024] f32, P[16][1024] bf16, Vt[64][128] bf16, reductions.
// ---------------------------------------------------------------------------
constexpr size_t ATTN_LDS =
    (size_t)16 * 1024 * 4 + (size_t)16 * 1024 * 2 + (size_t)64 * 128 * 2 + (128 + 32) * 4;

__global__ void attention_wmma(const __bf16* __restrict__ qkv,
                               const float* __restrict__ attn_size,
                               __bf16* __restrict__ xa) {
  extern __shared__ char smem[];
  float* S   = (float*)smem;                                   // 16*1024 f32
  __bf16* P  = (__bf16*)(smem + 16 * 1024 * 4);                // 16*1024 bf16
  __bf16* Vt = (__bf16*)(smem + 16 * 1024 * 4 + 16 * 1024 * 2);// 64*128 bf16
  float* red = (float*)(smem + 16 * 1024 * 4 + 16 * 1024 * 2 + 64 * 128 * 2);
  float* rmax = red + 128;
  float* rsum = rmax + 16;

  const int tid = threadIdx.x, lane = tid & 31, wv = tid >> 5;
  const int bh = blockIdx.x >> 6;   // / (N/16)
  const int tile = blockIdx.x & 63;
  const int b = bh / kH, hh = bh % kH;
  const int i0 = tile * 16;
  const float scale = 0.125f; // HD^-0.5

  // ---- phase 1: S = scale * Q_tile @ K^T, WMMA straight from global bf16
  const size_t qbase = ((size_t)b * kN + i0) * (3 * kC) + hh * kHD;
  for (int jt = wv; jt < 64; jt += 4) {           // uniform per wave
    const int jb = jt * 16;
    const size_t kbase = ((size_t)b * kN + jb) * (3 * kC) + kC + hh * kHD;
    floatx8 acc = {};
#pragma unroll
    for (int kk = 0; kk < 64; kk += 32) {
      bf16x16 af  = load_a_frag(qkv + qbase + kk, 3 * kC, lane);
      bf16x16 bfr = load_b_frag(qkv + kbase + kk, 3 * kC, lane);
      acc = wmma_bf16(af, bfr, acc);
    }
    const int hi = (lane >= 16) ? 8 : 0, nn = lane & 15;
#pragma unroll
    for (int v = 0; v < 8; ++v) S[(v + hi) * 1024 + jb + nn] = acc[v] * scale;
  }
  __syncthreads();

  // ---- phase 2: bias (+log attn_size[j]) and row softmax
  const int row = tid & 15, part = tid >> 4;      // 8 partials per row
  float mx = -3.4e38f;
  for (int j = part * 128; j < part * 128 + 128; ++j) {
    float vv = S[row * 1024 + j] + logf(attn_size[(size_t)b * kN + j]);
    S[row * 1024 + j] = vv;
    mx = fmaxf(mx, vv);
  }
  red[row * 8 + part] = mx; __syncthreads();
  if (tid < 16) {
    float m2 = red[tid * 8];
    for (int p = 1; p < 8; ++p) m2 = fmaxf(m2, red[tid * 8 + p]);
    rmax[tid] = m2;
  }
  __syncthreads();
  float sm = 0.f; const float rm = rmax[row];
  for (int j = part * 128; j < part * 128 + 128; ++j) {
    float e = expf(S[row * 1024 + j] - rm);
    S[row * 1024 + j] = e; sm += e;
  }
  red[row * 8 + part] = sm; __syncthreads();
  if (tid < 16) {
    float s2 = 0.f;
    for (int p = 0; p < 8; ++p) s2 += red[tid * 8 + p];
    rsum[tid] = s2;
  }
  __syncthreads();
  const float inv = 1.f / rsum[row];
  for (int j = part * 128; j < part * 128 + 128; ++j)
    P[row * 1024 + j] = (__bf16)(S[row * 1024 + j] * inv);
  __syncthreads();

  // ---- phase 3: out = P @ V, staging V transposed in LDS (j-contiguous rows)
  const int dbase = wv * 16;  // wave owns 16 head-dim columns
  floatx8 acc = {};
  for (int ch = 0; ch < 8; ++ch) {
    const int j = ch * 128 + tid;
    const __bf16* vrow = qkv + ((size_t)b * kN + j) * (3 * kC) + 2 * kC + hh * kHD;
#pragma unroll
    for (int d = 0; d < 64; ++d) Vt[d * 128 + tid] = vrow[d];
    __syncthreads();
#pragma unroll
    for (int jt = 0; jt < 4; ++jt) {
      const int jb = jt * 32;
      bf16x16 af  = load_a_frag(P + ch * 128 + jb, 1024, lane);
      bf16x16 bfr = load_b_frag(Vt + dbase * 128 + jb, 128, lane);
      acc = wmma_bf16(af, bfr, acc);
    }
    __syncthreads();
  }
  const int hi = (lane >= 16) ? 8 : 0, nn = lane & 15;
#pragma unroll
  for (int v = 0; v < 8; ++v)
    xa[((size_t)b * kN + i0 + v + hi) * kC + hh * kHD + dbase + nn] = (__bf16)acc[v];
}

// ---------------------------------------------------------------------------
// metric = normalize(mean_h(K)); one block (64 threads = head dim) per token.
// ---------------------------------------------------------------------------
__global__ void metric_norm(const __bf16* __restrict__ qkv, float* __restrict__ mvec) {
  const int bn = blockIdx.x;  // b*N + n
  const int t = threadIdx.x;  // d in [0,64)
  const size_t base = (size_t)bn * (3 * kC) + kC + t;
  float s = 0.f;
#pragma unroll
  for (int h = 0; h < kH; ++h) s += (float)qkv[base + h * kHD];
  s *= (1.f / (float)kH);
  __shared__ float sq[64];
  sq[t] = s * s; __syncthreads();
  for (int off = 32; off > 0; off >>= 1) {
    if (t < off) sq[t] += sq[t + off];
    __syncthreads();
  }
  mvec[(size_t)bn * kHD + t] = s * rsqrtf(sq[0]);
}

// ---------------------------------------------------------------------------
// scores[b,i,:] = a_i . b_j; keep row max + argmax (first-index tiebreak).
// One block (64 threads) per (b, i).
// ---------------------------------------------------------------------------
__global__ void tome_scores(const float* __restrict__ mvec,
                            float* __restrict__ node_max, int* __restrict__ node_idx) {
  const int b = blockIdx.x >> 9;
  const int i = blockIdx.x & 511;
  const int t = threadIdx.x;
  __shared__ float ac[64];
  ac[t] = mvec[((size_t)b * kN + 2 * i) * kHD + t];
  __syncthreads();
  float best = -3.4e38f; int bj = 0x7fffffff;
  for (int j = t; j < kN / 2; j += 64) {
    const float* br = mvec + ((size_t)b * kN + 2 * j + 1) * kHD;
    float d = 0.f;
#pragma unroll
    for (int c = 0; c < 64; ++c) d += ac[c] * br[c];
    if (d > best || (d == best && j < bj)) { best = d; bj = j; }
  }
  __shared__ float bk[64]; __shared__ int bi[64];
  bk[t] = best; bi[t] = bj; __syncthreads();
  for (int off = 32; off > 0; off >>= 1) {
    if (t < off) {
      if (bk[t + off] > bk[t] || (bk[t + off] == bk[t] && bi[t + off] < bi[t])) {
        bk[t] = bk[t + off]; bi[t] = bi[t + off];
      }
    }
    __syncthreads();
  }
  if (t == 0) { node_max[blockIdx.x] = bk[0]; node_idx[blockIdx.x] = bi[0]; }
}

// ---------------------------------------------------------------------------
// Per-batch stable descending argsort of node_max (512 keys, bitonic).
// Comparator (key desc, index asc) reproduces jnp.argsort(-node_max) (stable).
// ---------------------------------------------------------------------------
__global__ void tome_sort(const float* __restrict__ node_max, const int* __restrict__ node_idx,
                          int* __restrict__ src_idx, int* __restrict__ dst_idx,
                          int* __restrict__ unm_idx) {
  const int b = blockIdx.x, t = threadIdx.x;
  __shared__ float key[512];
  __shared__ int   idx[512];
  key[t] = node_max[(size_t)b * 512 + t];
  idx[t] = t;
  __syncthreads();
  for (int k = 2; k <= 512; k <<= 1) {
    for (int j = k >> 1; j > 0; j >>= 1) {
      const int p = t ^ j;
      if (p > t) {
        const bool up = ((t & k) == 0);
        const float ka = key[t], kb = key[p];
        const int ia = idx[t], ib = idx[p];
        const bool aFirst = (ka > kb) || (ka == kb && ia < ib); // descending, stable
        if (up ? !aFirst : aFirst) { key[t] = kb; key[p] = ka; idx[t] = ib; idx[p] = ia; }
      }
      __syncthreads();
    }
  }
  if (t < kR) {
    src_idx[(size_t)b * kR + t] = idx[t];
    dst_idx[(size_t)b * kR + t] = node_idx[(size_t)b * 512 + idx[t]];
  } else if (kNU > 0) {
    unm_idx[(size_t)b * (kNU > 0 ? kNU : 1) + (t - kR)] = idx[t];
  }
}

// ---------------------------------------------------------------------------
// Merge: init dst rows (odd tokens), then atomic scatter-add of src rows.
// ---------------------------------------------------------------------------
__global__ void merge_init(const float* __restrict__ xres, const float* __restrict__ asz,
                           float* __restrict__ xm, float* __restrict__ sz) {
  const int b = blockIdx.x >> 9, j = blockIdx.x & 511;
  const size_t srow = (size_t)b * kN + 2 * j + 1;
  const float a = asz[srow];
  const size_t orow = (size_t)b * kTD + kNU + j;
  for (int c = threadIdx.x; c < kC; c += 256)
    xm[orow * kC + c] = xres[srow * kC + c] * a;
  if (threadIdx.x == 0) sz[orow] = a;
}

__global__ void merge_unm(const float* __restrict__ xres, const float* __restrict__ asz,
                          const int* __restrict__ unm_idx,
                          float* __restrict__ xm, float* __restrict__ sz) {
  const int b = blockIdx.x / (kNU > 0 ? kNU : 1), u = blockIdx.x % (kNU > 0 ? kNU : 1);
  const int is = unm_idx[(size_t)b * (kNU > 0 ? kNU : 1) + u];
  const size_t srow = (size_t)b * kN + 2 * is;
  const float a = asz[srow];
  const size_t orow = (size_t)b * kTD + u;
  for (int c = threadIdx.x; c < kC; c += 256)
    xm[orow * kC + c] = xres[srow * kC + c] * a;
  if (threadIdx.x == 0) sz[orow] = a;
}

__global__ void merge_scatter(const float* __restrict__ xres, const float* __restrict__ asz,
                              const int* __restrict__ src_idx, const int* __restrict__ dst_idx,
                              float* __restrict__ xm, float* __restrict__ sz) {
  const int b = blockIdx.x >> 9, s = blockIdx.x & 511;
  const int is = src_idx[(size_t)b * kR + s];
  const int jd = dst_idx[(size_t)b * kR + s];
  const size_t srow = (size_t)b * kN + 2 * is;
  const float a = asz[srow];
  const size_t orow = (size_t)b * kTD + kNU + jd;
  for (int c = threadIdx.x; c < kC; c += 256)
    atomicAdd(&xm[orow * kC + c], xres[srow * kC + c] * a);
  if (threadIdx.x == 0) atomicAdd(&sz[orow], a);
}

__global__ void div_size(const float* __restrict__ xm, const float* __restrict__ sz,
                         float* __restrict__ x2) {
  const int rowt = blockIdx.x;
  const float s = sz[rowt];
  for (int c = threadIdx.x; c < kC; c += 256)
    x2[(size_t)rowt * kC + c] = xm[(size_t)rowt * kC + c] / s;
}

__global__ void copy_size(const float* __restrict__ sz, float* __restrict__ out, int n) {
  const int i = blockIdx.x * 256 + threadIdx.x;
  if (i < n) out[i] = sz[i];
}

// ---------------------------------------------------------------------------
// Host driver
// ---------------------------------------------------------------------------
extern "C" void kernel_launch(void* const* d_in, const int* in_sizes, int n_in,
                              void* d_out, int out_size, void* d_ws, size_t ws_size,
                              hipStream_t stream) {
  const float* x         = (const float*)d_in[0];
  const float* attn_size = (const float*)d_in[1];
  const float* ln1_w     = (const float*)d_in[2];
  const float* ln1_b     = (const float*)d_in[3];
  const float* qkv_w     = (const float*)d_in[4];
  const float* proj_w    = (const float*)d_in[5];
  const float* proj_b    = (const float*)d_in[6];
  const float* ln2_w     = (const float*)d_in[7];
  const float* ln2_b     = (const float*)d_in[8];
  const float* fc1_w     = (const float*)d_in[9];
  const float* fc1_b     = (const float*)d_in[10];
  const float* fc2_w     = (const float*)d_in[11];
  const float* fc2_b     = (const float*)d_in[12];
  float* out = (float*)d_out;

  char* wsp = (char*)d_ws;
  auto alloc = [&](size_t bytes) -> char* {
    char* p = wsp;
    wsp += (bytes + 255) & ~(size_t)255;
    return p;
  };

  const size_t M1 = (size_t)kB * kN;   // 8192 token rows pre-merge
  const size_t M2 = (size_t)kB * kTD;  // 4096 token rows post-merge

  __bf16* h16   = (__bf16*)alloc(M1 * kC * 2);
  __bf16* wq16  = (__bf16*)alloc((size_t)3 * kC * kC * 2);
  __bf16* wp16  = (__bf16*)alloc((size_t)kC * kC * 2);
  __bf16* wf116 = (__bf16*)alloc((size_t)4 * kC * kC * 2);
  __bf16* wf216 = (__bf16*)alloc((size_t)kC * 4 * kC * 2);
  __bf16* qkv16 = (__bf16*)alloc(M1 * 3 * kC * 2);
  __bf16* xa16  = (__bf16*)alloc(M1 * kC * 2);
  float*  xres  = (float*)alloc(M1 * kC * 4);
  float*  mvec  = (float*)alloc(M1 * kHD * 4);
  float*  nmax  = (float*)alloc((size_t)kB * 512 * 4);
  int*    nidx  = (int*)alloc((size_t)kB * 512 * 4);
  int*    sidx  = (int*)alloc((size_t)kB * kR * 4);
  int*    didx  = (int*)alloc((size_t)kB * kR * 4);
  int*    uidx  = (int*)alloc((size_t)kB * (kNU > 0 ? kNU : 1) * 4);
  float*  xm    = (float*)alloc(M2 * kC * 4);
  float*  sizew = (float*)alloc(M2 * 4);
  float*  x2    = (float*)alloc(M2 * kC * 4);
  __bf16* h216  = (__bf16*)alloc(M2 * kC * 2);
  __bf16* hm16  = (__bf16*)alloc(M2 * 4 * kC * 2);

  // --- weight conversion to bf16
  {
    int n;
    n = 3 * kC * kC; cvt_f32_to_bf16<<<(n + 255) / 256, 256, 0, stream>>>(qkv_w, wq16, n);
    n = kC * kC;     cvt_f32_to_bf16<<<(n + 255) / 256, 256, 0, stream>>>(proj_w, wp16, n);
    n = 4 * kC * kC; cvt_f32_to_bf16<<<(n + 255) / 256, 256, 0, stream>>>(fc1_w, wf116, n);
    n = kC * 4 * kC; cvt_f32_to_bf16<<<(n + 255) / 256, 256, 0, stream>>>(fc2_w, wf216, n);
  }

  // --- LN1 -> h (bf16)
  layernorm_to_bf16<<<(int)M1, 256, 0, stream>>>(x, ln1_w, ln1_b, h16, kC);

  // --- qkv = h @ qkv_w^T   (M=8192, N=2304, K=768), store bf16
  gemm_bf16_nt<<<dim3(3 * kC / 128, (int)M1 / 64), 128, 0, stream>>>(
      h16, wq16, nullptr, nullptr, nullptr, qkv16, (int)M1, 3 * kC, kC, 0);

  // --- attention (per b,h,16-query tile)
  attention_wmma<<<kB * kH * (kN / 16), 128, (int)ATTN_LDS, stream>>>(qkv16, attn_size, xa16);

  // --- metric = normalize(mean_h K)
  metric_norm<<<(int)M1, 64, 0, stream>>>(qkv16, mvec);

  // --- x = x + (xa @ proj_w^T + proj_b)   (fp32 out)
  gemm_bf16_nt<<<dim3(kC / 128, (int)M1 / 64), 128, 0, stream>>>(
      xa16, wp16, proj_b, x, xres, nullptr, (int)M1, kC, kC, 1);

  // --- ToMe matching
  tome_scores<<<kB * 512, 64, 0, stream>>>(mvec, nmax, nidx);
  tome_sort<<<kB, 512, 0, stream>>>(nmax, nidx, sidx, didx, uidx);

  // --- merge (x * attn_size and attn_size)
  merge_init<<<kB * 512, 256, 0, stream>>>(xres, attn_size, xm, sizew);
  if (kNU > 0)
    merge_unm<<<kB * (kNU > 0 ? kNU : 1), 256, 0, stream>>>(xres, attn_size, uidx, xm, sizew);
  merge_scatter<<<kB * 512, 256, 0, stream>>>(xres, attn_size, sidx, didx, xm, sizew);
  div_size<<<(int)M2, 256, 0, stream>>>(xm, sizew, x2);

  // --- LN2 -> h2 (bf16)
  layernorm_to_bf16<<<(int)M2, 256, 0, stream>>>(x2, ln2_w, ln2_b, h216, kC);

  // --- fc1 + GELU  (M=4096, N=3072, K=768), bf16 out
  gemm_bf16_nt<<<dim3(4 * kC / 128, (int)M2 / 64), 128, 0, stream>>>(
      h216, wf116, fc1_b, nullptr, nullptr, hm16, (int)M2, 4 * kC, kC, 2);

  // --- x_out = x2 + (hm @ fc2_w^T + fc2_b) -> d_out (first output)
  gemm_bf16_nt<<<dim3(kC / 128, (int)M2 / 64), 128, 0, stream>>>(
      hm16, wf216, fc2_b, x2, out, nullptr, (int)M2, kC, 4 * kC, 1);

  // --- size -> d_out tail (second output)
  copy_size<<<((int)M2 + 255) / 256, 256, 0, stream>>>(sizew, out + M2 * kC, (int)M2);
}
